// ODGAT_36867999268896
// MI455X (gfx1250) — compile-verified, matrix operand order checked by reading, not run on hardware
//
#include <hip/hip_runtime.h>
#include <hip/hip_bf16.h>

typedef __attribute__((ext_vector_type(16))) _Float16 v16h;
typedef __attribute__((ext_vector_type(8)))  _Float16 v8h;
typedef __attribute__((ext_vector_type(8)))  float    v8f;

#define Bsz 32
#define Nn  512
#define INF_ 64
#define C1  256   // H*HID
#define Hh  4
#define HID 64
#define OUTF 64

union Frag16 { v16h v; v8h h[2]; };

// Load one 16-bit WMMA A/B fragment from a K-contiguous row of LDS.
// base = start of this row's K-window (32 K values); g = lane>>4.
__device__ __forceinline__ v16h ldfrag(const _Float16* base, int g) {
    Frag16 f;
    f.h[0] = *(const v8h*)(base + 8 * g);        // K = 8g .. 8g+7
    f.h[1] = *(const v8h*)(base + 16 + 8 * g);   // K = 16+8g .. 16+8g+7
    return f.v;
}

__device__ __forceinline__ v8h pack8(v8f a) {
    v8h o;
#pragma unroll
    for (int i = 0; i < 8; ++i) o[i] = (_Float16)a[i];
    return o;
}

// ---------------- Layer-1 projection: H1t[c][s] = f16(x @ W1)^T ----------------
__global__ __launch_bounds__(128) void proj1_kernel(
    const float* __restrict__ x,   // [B,512,64]
    const float* __restrict__ W1,  // [64,256]
    _Float16* __restrict__ H1t)    // [B,256,512] f16 (transposed)
{
    __shared__ __align__(16) _Float16 sx[16 * 64];    // A tile [m][k]
    __shared__ __align__(16) _Float16 swT[256 * 64];  // B^T [col][k]
    int blk = blockIdx.x;
    int b = blk >> 5, rt = blk & 31, n0 = rt * 16;
    const float* xg = x + ((size_t)b * Nn + n0) * INF_;
    {   // stage x tile: 1024 f32 -> f16, vectorized
        const float4* xg4 = (const float4*)xg;
        float4 u = xg4[threadIdx.x * 2], w = xg4[threadIdx.x * 2 + 1];
        v8h t;
        t[0] = (_Float16)u.x; t[1] = (_Float16)u.y; t[2] = (_Float16)u.z; t[3] = (_Float16)u.w;
        t[4] = (_Float16)w.x; t[5] = (_Float16)w.y; t[6] = (_Float16)w.z; t[7] = (_Float16)w.w;
        *(v8h*)(sx + threadIdx.x * 8) = t;
    }
    for (int c = threadIdx.x; c < 256; c += 128) {    // stage W1 transposed
        for (int k0 = 0; k0 < 64; k0 += 8) {
            v8h t;
#pragma unroll
            for (int j = 0; j < 8; ++j) t[j] = (_Float16)W1[(size_t)(k0 + j) * 256 + c];
            *(v8h*)(swT + c * 64 + k0) = t;
        }
    }
    __syncthreads();

    int lane = threadIdx.x & 31, w = threadIdx.x >> 5;
    int r = lane & 15, g = lane >> 4;
    v16h a0 = ldfrag(sx + r * 64, g);
    v16h a1 = ldfrag(sx + r * 64 + 32, g);
    _Float16* outg = H1t + (size_t)b * C1 * Nn;
#pragma unroll
    for (int ct = 0; ct < 4; ++ct) {
        int c0 = w * 64 + ct * 16;
        v16h b0 = ldfrag(swT + (c0 + r) * 64, g);
        v16h b1 = ldfrag(swT + (c0 + r) * 64 + 32, g);
        v8f acc = {};
        acc = __builtin_amdgcn_wmma_f32_16x16x32_f16(false, a0, false, b0, (short)0, acc, false, false);
        acc = __builtin_amdgcn_wmma_f32_16x16x32_f16(false, a1, false, b1, (short)0, acc, false, false);
        // D: lane column = c0+r, rows m = i+8g contiguous -> one b128 store
        *(v8h*)(outg + (size_t)(c0 + r) * Nn + n0 + 8 * g) = pack8(acc);
    }
}

// -------- e-scores from transposed H[c][s]: coalesced over nodes --------
__global__ __launch_bounds__(128) void escore_kernel(
    const _Float16* __restrict__ Hm,  // [B, heads*feat, 512]
    const float* __restrict__ a_src,  // [heads*feat]
    const float* __restrict__ a_dst,
    float* __restrict__ es, float* __restrict__ ed,
    int heads, int feat)
{
    int node = blockIdx.x * 128 + threadIdx.x;   // 0..B*N-1
    int b = node >> 9, n = node & (Nn - 1);
    int C = heads * feat;
    const _Float16* hb = Hm + (size_t)b * C * Nn;
    for (int h = 0; h < heads; ++h) {
        float ss = 0.f, sd = 0.f;
        for (int f = 0; f < feat; ++f) {
            float hv = (float)hb[(size_t)(h * feat + f) * Nn + n];
            ss += hv * a_src[h * feat + f];
            sd += hv * a_dst[h * feat + f];
        }
        es[(size_t)node * heads + h] = ss;
        ed[(size_t)node * heads + h] = sd;
    }
}

// ------------- Layer-1 attention (fused softmax+aggregate+bias+ELU) -------------
__global__ __launch_bounds__(128) void attn1_kernel(
    const int* __restrict__ adj,        // [B,512,512]
    const _Float16* __restrict__ H1t,   // [B,256,512]
    const float* __restrict__ es,       // [B*512,4]
    const float* __restrict__ ed,       // [B*512,4]
    const float* __restrict__ b1,       // [256]
    _Float16* __restrict__ h1a)         // [B,512,256] row-major (A of proj2)
{
    __shared__ unsigned smask[16 * 16];               // mask bits [d][word]
    __shared__ __align__(16) float ses[Nn * Hh];      // e_src staged
    __shared__ __align__(16) _Float16 stile[C1 * 32]; // B^T tile [c][k]
    int blk = blockIdx.x;
    int b = blk >> 5, dt = blk & 31, d0 = dt * 16;
    const int* adjb = adj + (size_t)b * Nn * Nn;

    for (int s = threadIdx.x; s < Nn; s += 128)       // warm L2/WGP$ for adj column reads
        __builtin_prefetch(adjb + (size_t)s * Nn + d0, 0, 1);

    for (int i = threadIdx.x; i < 256; i += 128) {    // build mask[d, s]
        int d = i >> 4, wd = i & 15, dg = d0 + d;
        unsigned bits = 0;
        for (int j = 0; j < 32; ++j) {
            int s = wd * 32 + j;
            int m = (adjb[(size_t)s * Nn + dg] != 0) | (s == dg);
            bits |= (unsigned)m << j;
        }
        smask[i] = bits;
    }
    for (int i = threadIdx.x; i < Nn * Hh; i += 128) ses[i] = es[(size_t)b * Nn * Hh + i];
    __syncthreads();

    int lane = threadIdx.x & 31, h = threadIdx.x >> 5;  // wave == head
    int r = lane & 15, g = lane >> 4;
    float edv = ed[((size_t)b * Nn + d0 + r) * Hh + h];

    float rmax = -1e30f;                         // pass 1a: row max
    for (int s = g * 256; s < g * 256 + 256; ++s) {
        bool mk = (smask[r * 16 + (s >> 5)] >> (s & 31)) & 1u;
        float sc = edv + ses[s * Hh + h];
        sc = sc > 0.f ? sc : 0.2f * sc;
        sc = mk ? sc : -1e9f;
        rmax = fmaxf(rmax, sc);
    }
    rmax = fmaxf(rmax, __shfl_xor(rmax, 16, 32));
    float rsum = 0.f;                            // pass 1b: row sum
    for (int s = g * 256; s < g * 256 + 256; ++s) {
        bool mk = (smask[r * 16 + (s >> 5)] >> (s & 31)) & 1u;
        float sc = edv + ses[s * Hh + h];
        sc = sc > 0.f ? sc : 0.2f * sc;
        sc = mk ? sc : -1e9f;
        rsum += __expf(sc - rmax);
    }
    rsum += __shfl_xor(rsum, 16, 32);
    float rinv = 1.f / rsum;

    v8f acc[4] = {};                             // pass 2: aggregate via WMMA
    const _Float16* H1b = H1t + (size_t)b * C1 * Nn;
    for (int s0 = 0; s0 < Nn; s0 += 32) {
        __syncthreads();
        // stage B^T tile [c][k]: K contiguous in H1t -> pure float4 copy
        for (int i = threadIdx.x; i < C1 * 4; i += 128) {
            int c = i >> 2, q = i & 3;
            ((float4*)(stile + c * 32))[q] =
                ((const float4*)(H1b + (size_t)c * Nn + s0))[q];
        }
        __syncthreads();
        v16h af;                                  // alpha fragment (row r, K=s)
#pragma unroll
        for (int e = 0; e < 16; ++e) {
            int k = 16 * (e >> 3) + 8 * g + (e & 7);
            int s = s0 + k;
            bool mk = (smask[r * 16 + (s >> 5)] >> (s & 31)) & 1u;
            float sc = edv + ses[s * Hh + h];
            sc = sc > 0.f ? sc : 0.2f * sc;
            float al = mk ? __expf(sc - rmax) * rinv : 0.f;
            af[e] = (_Float16)al;
        }
#pragma unroll
        for (int ct = 0; ct < 4; ++ct) {
            int c0 = h * 64 + ct * 16;
            v16h bf = ldfrag(stile + (c0 + r) * 32, g);
            acc[ct] = __builtin_amdgcn_wmma_f32_16x16x32_f16(false, af, false, bf, (short)0, acc[ct], false, false);
        }
    }
    _Float16* outg = h1a + ((size_t)b * Nn + d0) * C1;
#pragma unroll
    for (int ct = 0; ct < 4; ++ct) {
        int c0 = h * 64 + ct * 16;
        float bias = b1[c0 + r];
#pragma unroll
        for (int i = 0; i < 8; ++i) {
            float vv = acc[ct][i] + bias;
            vv = vv > 0.f ? vv : (__expf(vv) - 1.f);   // ELU
            outg[(size_t)(i + 8 * g) * C1 + c0 + r] = (_Float16)vv;
        }
    }
}

// ---------------- Layer-2 projection: H2t[c][s] = f16(h1a @ W2)^T ----------------
__global__ __launch_bounds__(128) void proj2_kernel(
    const _Float16* __restrict__ h1a, // [B,512,256] row-major
    const float* __restrict__ W2,     // [256,64]
    _Float16* __restrict__ H2t)       // [B,64,512] transposed
{
    __shared__ __align__(16) _Float16 sa[16 * 256];   // A [m][k]
    __shared__ __align__(16) _Float16 swT[64 * 256];  // B^T [col][k]
    int blk = blockIdx.x;
    int b = blk >> 5, rt = blk & 31, n0 = rt * 16;
    const _Float16* ag = h1a + ((size_t)b * Nn + n0) * C1;
    for (int i = threadIdx.x; i < 16 * 256 / 8; i += 128)
        ((float4*)sa)[i] = ((const float4*)ag)[i];
    {
        int c = threadIdx.x & 63, kh = threadIdx.x >> 6;  // 2 threads per col
        for (int k0 = kh * 128; k0 < kh * 128 + 128; k0 += 8) {
            v8h t;
#pragma unroll
            for (int j = 0; j < 8; ++j) t[j] = (_Float16)W2[(size_t)(k0 + j) * 64 + c];
            *(v8h*)(swT + c * 256 + k0) = t;
        }
    }
    __syncthreads();

    int lane = threadIdx.x & 31, w = threadIdx.x >> 5;
    int r = lane & 15, g = lane >> 4;
    int c0 = w * 16;
    v8f acc = {};
    for (int k0 = 0; k0 < 256; k0 += 32) {
        v16h af = ldfrag(sa + r * 256 + k0, g);
        v16h bf = ldfrag(swT + (c0 + r) * 256 + k0, g);
        acc = __builtin_amdgcn_wmma_f32_16x16x32_f16(false, af, false, bf, (short)0, acc, false, false);
    }
    _Float16* outg = H2t + (size_t)b * OUTF * Nn;
    *(v8h*)(outg + (size_t)(c0 + r) * Nn + n0 + 8 * g) = pack8(acc);
}

// ------------- Layer-2 attention (1 head) -> final fp32 output -------------
__global__ __launch_bounds__(128) void attn2_kernel(
    const int* __restrict__ adj,
    const _Float16* __restrict__ H2t, // [B,64,512]
    const float* __restrict__ es,     // [B*512]
    const float* __restrict__ ed,
    const float* __restrict__ b2,     // [64]
    float* __restrict__ out)          // [B,512,64]
{
    __shared__ unsigned smask[16 * 16];
    __shared__ __align__(16) float ses[Nn];
    __shared__ __align__(16) _Float16 stile[OUTF * 32]; // [c][k]
    int blk = blockIdx.x;
    int b = blk >> 5, dt = blk & 31, d0 = dt * 16;
    const int* adjb = adj + (size_t)b * Nn * Nn;

    for (int s = threadIdx.x; s < Nn; s += 128)
        __builtin_prefetch(adjb + (size_t)s * Nn + d0, 0, 1);

    for (int i = threadIdx.x; i < 256; i += 128) {
        int d = i >> 4, wd = i & 15, dg = d0 + d;
        unsigned bits = 0;
        for (int j = 0; j < 32; ++j) {
            int s = wd * 32 + j;
            int m = (adjb[(size_t)s * Nn + dg] != 0) | (s == dg);
            bits |= (unsigned)m << j;
        }
        smask[i] = bits;
    }
    for (int i = threadIdx.x; i < Nn; i += 128) ses[i] = es[(size_t)b * Nn + i];
    __syncthreads();

    int lane = threadIdx.x & 31, w = threadIdx.x >> 5;  // wave == col tile
    int r = lane & 15, g = lane >> 4;
    float edv = ed[(size_t)b * Nn + d0 + r];

    float rmax = -1e30f;
    for (int s = g * 256; s < g * 256 + 256; ++s) {
        bool mk = (smask[r * 16 + (s >> 5)] >> (s & 31)) & 1u;
        float sc = edv + ses[s];
        sc = sc > 0.f ? sc : 0.2f * sc;
        sc = mk ? sc : -1e9f;
        rmax = fmaxf(rmax, sc);
    }
    rmax = fmaxf(rmax, __shfl_xor(rmax, 16, 32));
    float rsum = 0.f;
    for (int s = g * 256; s < g * 256 + 256; ++s) {
        bool mk = (smask[r * 16 + (s >> 5)] >> (s & 31)) & 1u;
        float sc = edv + ses[s];
        sc = sc > 0.f ? sc : 0.2f * sc;
        sc = mk ? sc : -1e9f;
        rsum += __expf(sc - rmax);
    }
    rsum += __shfl_xor(rsum, 16, 32);
    float rinv = 1.f / rsum;

    v8f acc = {};
    const _Float16* H2b = H2t + (size_t)b * OUTF * Nn;
    int c0 = w * 16;
    for (int s0 = 0; s0 < Nn; s0 += 32) {
        __syncthreads();
        for (int i = threadIdx.x; i < OUTF * 4; i += 128) {
            int c = i >> 2, q = i & 3;
            ((float4*)(stile + c * 32))[q] =
                ((const float4*)(H2b + (size_t)c * Nn + s0))[q];
        }
        __syncthreads();
        v16h af;
#pragma unroll
        for (int e = 0; e < 16; ++e) {
            int k = 16 * (e >> 3) + 8 * g + (e & 7);
            int s = s0 + k;
            bool mk = (smask[r * 16 + (s >> 5)] >> (s & 31)) & 1u;
            float sc = edv + ses[s];
            sc = sc > 0.f ? sc : 0.2f * sc;
            float al = mk ? __expf(sc - rmax) * rinv : 0.f;
            af[e] = (_Float16)al;
        }
        v16h bf = ldfrag(stile + (c0 + r) * 32, g);
        acc = __builtin_amdgcn_wmma_f32_16x16x32_f16(false, af, false, bf, (short)0, acc, false, false);
    }
    float* outg = out + ((size_t)b * Nn + d0) * OUTF;
    float bias = b2[c0 + r];
#pragma unroll
    for (int i = 0; i < 8; ++i)
        outg[(size_t)(i + 8 * g) * OUTF + c0 + r] = acc[i] + bias;
}

extern "C" void kernel_launch(void* const* d_in, const int* in_sizes, int n_in,
                              void* d_out, int out_size, void* d_ws, size_t ws_size,
                              hipStream_t stream) {
    const float* x      = (const float*)d_in[0];
    const int*   adj    = (const int*)d_in[1];
    const float* W1     = (const float*)d_in[2];
    const float* a1_src = (const float*)d_in[3];
    const float* a1_dst = (const float*)d_in[4];
    const float* b1     = (const float*)d_in[5];
    const float* W2     = (const float*)d_in[6];
    const float* a2_src = (const float*)d_in[7];
    const float* a2_dst = (const float*)d_in[8];
    const float* b2     = (const float*)d_in[9];
    float* out = (float*)d_out;

    char* ws = (char*)d_ws;
    _Float16* H1t = (_Float16*)(ws);                       // 8 MB  [B,256,512]
    _Float16* h1a = (_Float16*)(ws + (8u << 20));          // 8 MB  [B,512,256]
    _Float16* H2t = (_Float16*)(ws + (16u << 20));         // 2 MB  [B,64,512]
    float* e1s = (float*)(ws + (18u << 20));               // 256 KB
    float* e1d = (float*)(ws + (18u << 20) + (256u << 10));
    float* e2s = (float*)(ws + (19u << 20));               // 64 KB
    float* e2d = (float*)(ws + (19u << 20) + (64u << 10));

    dim3 blk(128);
    proj1_kernel<<<Bsz * (Nn / 16), blk, 0, stream>>>(x, W1, H1t);
    escore_kernel<<<Bsz * Nn / 128, blk, 0, stream>>>(H1t, a1_src, a1_dst, e1s, e1d, Hh, HID);
    attn1_kernel<<<Bsz * (Nn / 16), blk, 0, stream>>>(adj, H1t, e1s, e1d, b1, h1a);
    proj2_kernel<<<Bsz * (Nn / 16), blk, 0, stream>>>(h1a, W2, H2t);
    escore_kernel<<<Bsz * Nn / 128, blk, 0, stream>>>(H2t, a2_src, a2_dst, e2s, e2d, 1, OUTF);
    attn2_kernel<<<Bsz * (Nn / 16), blk, 0, stream>>>(adj, H2t, e2s, e2d, b2, out);
}
